// Model_29944511987746
// MI455X (gfx1250) — matrix operand order repaired and checked by reading, not compiled, not run on hardware
//
#include <hip/hip_runtime.h>
#include <hip/hip_bf16.h>

typedef __bf16 bf16_t;
typedef __attribute__((ext_vector_type(16))) __bf16 v16bf;
typedef __attribute__((ext_vector_type(8)))  __bf16 v8bf;
typedef __attribute__((ext_vector_type(4)))  __bf16 v4bf;
typedef __attribute__((ext_vector_type(8)))  float  v8f;
typedef __attribute__((ext_vector_type(4)))  float  v4fn;

#define MROWS 128     // batch rows (always 128)
#define TK    32      // K chunk per LDS stage
#define LDSS  40      // padded LDS row stride in bf16 (80B: 16B-aligned, conflict-free)

union Frag { v16bf v; v8bf h[2]; };

static __device__ __forceinline__ v8f wmma_bf16(const Frag& a, const Frag& b, v8f c) {
    return __builtin_amdgcn_wmma_f32_16x16x32_bf16(
        /*neg_a=*/false, a.v, /*neg_b=*/false, b.v,
        /*c_mod=*/(short)0, c, /*reuse_a=*/false, /*reuse_b=*/false);
}

// Split a float4 into packed bf16 hi / lo residual vectors (bf16x3 decomposition).
static __device__ __forceinline__ void split4(const v4fn f, v4bf& hi, v4bf& lo) {
#pragma unroll
    for (int e = 0; e < 4; ++e) {
        const bf16_t h = (bf16_t)f[e];
        hi[e] = h;
        lo[e] = (bf16_t)(f[e] - (float)h);
    }
}

// C[128 x N] = A[128 x K] * W[K x N] + bias, optional ReLU.
// f32 streamed once from HBM (W non-temporal), split to bf16 hi/lo in LDS,
// bf16x3 WMMA with register prefetch of the next K-chunk behind the WMMA burst.
template<bool RELU>
__global__ __launch_bounds__(256)
void gemm_bf16x3(const float* __restrict__ A, const float* __restrict__ W,
                 const float* __restrict__ bias, float* __restrict__ out,
                 int K, int N)
{
    __shared__ __align__(16) bf16_t sAh[MROWS * LDSS];
    __shared__ __align__(16) bf16_t sAl[MROWS * LDSS];
    __shared__ __align__(16) bf16_t sWh[128 * LDSS];
    __shared__ __align__(16) bf16_t sWl[128 * LDSS];

    const int t    = threadIdx.x;
    const int lane = t & 31;
    const int wave = t >> 5;
    const int n0   = blockIdx.x * 128;

    v8f acc[8];
#pragma unroll
    for (int s = 0; s < 8; ++s) acc[s] = (v8f){0.f,0.f,0.f,0.f,0.f,0.f,0.f,0.f};

    // A stage: thread -> (row, 16-col half). W stage: thread -> 4k x 4c micro-tile.
    const int arow = t >> 1, acb = (t & 1) * 16;
    const int wk0  = (t & 7) * 4;          // 8 k-groups of 4 -> K=32
    const int wc0  = (t >> 3) * 4;         // 32 c-groups of 4 -> 128 cols

    v4fn aReg[4], wReg[4];

    auto load_chunk = [&](int kc) {
#pragma unroll
        for (int i = 0; i < 4; ++i)
            aReg[i] = *(const v4fn*)(A + (size_t)arow * K + kc + acb + 4 * i);
#pragma unroll
        for (int kk = 0; kk < 4; ++kk)     // W used exactly once -> non-temporal
            wReg[kk] = __builtin_nontemporal_load(
                (const v4fn*)(W + (size_t)(kc + wk0 + kk) * N + n0 + wc0));
    };

    load_chunk(0);

    for (int kc = 0; kc < K; kc += TK) {
        // ---- store current chunk to LDS as packed bf16 hi/lo (ds_store_b64) ----
#pragma unroll
        for (int i = 0; i < 4; ++i) {
            v4bf hi, lo;
            split4(aReg[i], hi, lo);
            *(v4bf*)&sAh[arow * LDSS + acb + 4 * i] = hi;
            *(v4bf*)&sAl[arow * LDSS + acb + 4 * i] = lo;
        }
        {
            v4bf whi[4], wlo[4];
#pragma unroll
            for (int kk = 0; kk < 4; ++kk) split4(wReg[kk], whi[kk], wlo[kk]);
            // transpose 4x4 in registers; store 4 consecutive k per column
#pragma unroll
            for (int cc = 0; cc < 4; ++cc) {
                v4bf th, tl;
#pragma unroll
                for (int kk = 0; kk < 4; ++kk) { th[kk] = whi[kk][cc]; tl[kk] = wlo[kk][cc]; }
                *(v4bf*)&sWh[(wc0 + cc) * LDSS + wk0] = th;
                *(v4bf*)&sWl[(wc0 + cc) * LDSS + wk0] = tl;
            }
        }
        __syncthreads();

        // ---- prefetch next chunk's global data behind the WMMA burst ----
        if (kc + TK < K) load_chunk(kc + TK);

        // ---- A fragments for this wave (rows wave*16..+15) ----
        // lanes 0-15: K0..15 ; lanes 16-31: K16..31
        const int ar   = wave * 16 + (lane & 15);
        const int aoff = (lane < 16) ? 0 : 16;
        Frag ah, al;
        ah.h[0] = *(const v8bf*)&sAh[ar * LDSS + aoff];
        ah.h[1] = *(const v8bf*)&sAh[ar * LDSS + aoff + 8];
        al.h[0] = *(const v8bf*)&sAl[ar * LDSS + aoff];
        al.h[1] = *(const v8bf*)&sAl[ar * LDSS + aoff + 8];

        // ---- 8 N-subtiles: B frag = column(lane&15), K0..15 (lanes<16) / K16..31 ----
#pragma unroll
        for (int s = 0; s < 8; ++s) {
            const int bc   = s * 16 + (lane & 15);
            const int boff = (lane < 16) ? 0 : 16;
            Frag bh, bl;
            bh.h[0] = *(const v8bf*)&sWh[bc * LDSS + boff];
            bh.h[1] = *(const v8bf*)&sWh[bc * LDSS + boff + 8];
            bl.h[0] = *(const v8bf*)&sWl[bc * LDSS + boff];
            bl.h[1] = *(const v8bf*)&sWl[bc * LDSS + boff + 8];
            acc[s] = wmma_bf16(ah, bh, acc[s]);   // hi*hi
            acc[s] = wmma_bf16(ah, bl, acc[s]);   // hi*lo
            acc[s] = wmma_bf16(al, bh, acc[s]);   // lo*hi
        }
        __syncthreads();
    }

    // ---- epilogue: bias (+relu), f32 out. C layout: VGPR r -> M = wave*16 + r + (lane<16?0:8)
    const int mbase = wave * 16 + ((lane < 16) ? 0 : 8);
#pragma unroll
    for (int s = 0; s < 8; ++s) {
        const int gn = n0 + s * 16 + (lane & 15);
        const float bv = bias[gn];
#pragma unroll
        for (int r = 0; r < 8; ++r) {
            float v = acc[s][r] + bv;
            if (RELU) v = fmaxf(v, 0.0f);
            out[(size_t)(mbase + r) * N + gn] = v;
        }
    }
}

// One block per (row, net). 1024 threads = 32 waves; wave w covers columns
// [1024w, 1024w+1024) so the per-wave max is exactly the per-x-block max.
__global__ __launch_bounds__(1024)
void softmax_stats(const float* __restrict__ logits,
                   float* __restrict__ Ma, float* __restrict__ invZa,
                   float* __restrict__ m2)
{
    const int row = blockIdx.x, net = blockIdx.y;
    const int t = threadIdx.x, lane = t & 31, wave = t >> 5;
    const float* __restrict__ L = logits + ((size_t)net * 128 + row) * 32768;

    __shared__ float smax[32], ssum[32];

    const float4* p = (const float4*)(L + t * 32);
    float4 v[8];
    float m = -3.402823466e38f;
#pragma unroll
    for (int i = 0; i < 8; ++i) {
        v[i] = p[i];
        m = fmaxf(m, fmaxf(fmaxf(v[i].x, v[i].y), fmaxf(v[i].z, v[i].w)));
    }
#pragma unroll
    for (int off = 16; off > 0; off >>= 1) m = fmaxf(m, __shfl_xor(m, off));
    if (lane == 0) smax[wave] = m;
    __syncthreads();

    float rmax = -3.402823466e38f;
#pragma unroll
    for (int i = 0; i < 32; ++i) rmax = fmaxf(rmax, smax[i]);

    float s = 0.0f;
#pragma unroll
    for (int i = 0; i < 8; ++i) {
        s += __expf(v[i].x - rmax) + __expf(v[i].y - rmax)
           + __expf(v[i].z - rmax) + __expf(v[i].w - rmax);
    }
#pragma unroll
    for (int off = 16; off > 0; off >>= 1) s += __shfl_xor(s, off);
    if (lane == 0) ssum[wave] = s;
    __syncthreads();

    float tot = 0.0f;
#pragma unroll
    for (int i = 0; i < 32; ++i) tot += ssum[i];

    if (net == 0) {
        if (t == 0) { Ma[row] = rmax; invZa[row] = 1.0f / tot; }
    } else {
        // m2[row][x] = max softmax prob in block x = exp(blockmax - rowmax)/Z
        if (lane == 0) m2[row * 32 + wave] = __expf(smax[wave] - rmax) / tot;
    }
}

// out[B,j] = max_x min( exp(la[B, x*1024+j]-Ma)/Za , m2[B,x] )
__global__ __launch_bounds__(256)
void finalize_join(const float* __restrict__ la, const float* __restrict__ Ma,
                   const float* __restrict__ invZa, const float* __restrict__ m2,
                   float* __restrict__ out)
{
    const int g = blockIdx.x * blockDim.x + threadIdx.x;   // 0..131071
    const int row = g >> 10, j = g & 1023;
    const float M  = Ma[row];
    const float iz = invZa[row];
    const float* __restrict__ Lr  = la + (size_t)row * 32768 + j;
    const float* __restrict__ m2r = m2 + row * 32;
    float r = 0.0f;
#pragma unroll
    for (int x = 0; x < 32; ++x) {
        const float p = __expf(Lr[x << 10] - M) * iz;
        r = fmaxf(r, fminf(p, m2r[x]));
    }
    out[g] = r;
}

extern "C" void kernel_launch(void* const* d_in, const int* in_sizes, int n_in,
                              void* d_out, int out_size, void* d_ws, size_t ws_size,
                              hipStream_t stream) {
    (void)in_sizes; (void)n_in; (void)out_size; (void)ws_size;
    const float* x   = (const float*)d_in[0];
    const float* W1a = (const float*)d_in[1];
    const float* b1a = (const float*)d_in[2];
    const float* W2a = (const float*)d_in[3];
    const float* b2a = (const float*)d_in[4];
    const float* W1b = (const float*)d_in[5];
    const float* b1b = (const float*)d_in[6];
    const float* W2b = (const float*)d_in[7];
    const float* b2b = (const float*)d_in[8];
    float* out = (float*)d_out;

    float* ws     = (float*)d_ws;
    float* h      = ws;                      // 2 * 128*1024
    float* logits = ws + 2 * 128 * 1024;     // 2 * 128*32768
    float* Ma     = logits + 2 * 128 * 32768;
    float* invZa  = Ma + 128;
    float* m2     = invZa + 128;             // 128*32

    // GEMM1 + bias + relu (both nets)
    gemm_bf16x3<true ><<<dim3(1024 / 128), 256, 0, stream>>>(x, W1a, b1a, h,              1024, 1024);
    gemm_bf16x3<true ><<<dim3(1024 / 128), 256, 0, stream>>>(x, W1b, b1b, h + 128 * 1024, 1024, 1024);
    // GEMM2 + bias -> logits (both nets); W2 (268MB) streamed exactly once, NT
    gemm_bf16x3<false><<<dim3(32768 / 128), 256, 0, stream>>>(h,              W2a, b2a, logits,                 1024, 32768);
    gemm_bf16x3<false><<<dim3(32768 / 128), 256, 0, stream>>>(h + 128 * 1024, W2b, b2b, logits + 128 * 32768, 1024, 32768);
    // softmax stats: rowmax / 1/Z for net a, block maxima -> m2 for net b
    softmax_stats<<<dim3(128, 2), 1024, 0, stream>>>(logits, Ma, invZa, m2);
    // join
    finalize_join<<<dim3(128 * 1024 / 256), 256, 0, stream>>>(logits, Ma, invZa, m2, out);
}
